// SelfAttention_27891517620330
// MI455X (gfx1250) — matrix operand order repaired and checked by reading, not compiled, not run on hardware
//
#include <hip/hip_runtime.h>

typedef __attribute__((ext_vector_type(16))) _Float16 v16h;
typedef __attribute__((ext_vector_type(8)))  _Float16 v8h;
typedef __attribute__((ext_vector_type(8)))  float    v8f;
typedef int v4i_t __attribute__((vector_size(16)));   // matches builtin signature

#define EMBED 128
#define SEQ   4096
#define BATCH 4
#define SCALE 0.08838834764831845f   // 1/sqrt(128)

#if defined(__has_builtin)
#if __has_builtin(__builtin_amdgcn_global_load_async_to_lds_b128)
#define HAVE_ASYNC_LDS 1
#endif
#endif

__device__ __forceinline__ v8f wmma16(v16h a, v16h b, v8f c) {
    // D = A(16x32 f16) * B(32x16 f16) + C(16x16 f32)
    return __builtin_amdgcn_wmma_f32_16x16x32_f16(false, a, false, b,
                                                  (short)0, c, false, false);
}

// 16-byte global -> LDS copy chunk: async on CDNA5, sync fallback otherwise.
__device__ __forceinline__ void copy16(const void* g, void* l) {
#ifdef HAVE_ASYNC_LDS
    __builtin_amdgcn_global_load_async_to_lds_b128(
        (__attribute__((address_space(1))) v4i_t*)g,
        (__attribute__((address_space(3))) v4i_t*)l, 0, 0);
#else
    *(v8h*)l = *(const v8h*)g;
#endif
}

// ---------------------------------------------------------------------------
// Kernel 1: QKV projection.  One wave computes a 16-row tile of Q, K, V.
//   Q,K stored row-major f16; V stored transposed Vt[b][d][s].
// ---------------------------------------------------------------------------
__device__ __forceinline__ v8f proj_acc(const float* __restrict__ W,
                                        const v16h a[4], int nt, int n, int kh) {
    v8f acc = {};
#pragma unroll
    for (int ks = 0; ks < 4; ++ks) {
        // B column n = row (nt*16+n) of W over d : contiguous
        const float* wp = W + (size_t)(nt * 16 + n) * EMBED + ks * 32 + kh;
        v16h bm;
#pragma unroll
        for (int j = 0; j < 16; ++j) bm[j] = (_Float16)wp[j];
        acc = wmma16(a[ks], bm, acc);
    }
    return acc;
}

__global__ __launch_bounds__(256) void qkv_proj_kernel(
    const float* __restrict__ x,
    const float* __restrict__ Wq, const float* __restrict__ bq,
    const float* __restrict__ Wk, const float* __restrict__ bk,
    const float* __restrict__ Wv, const float* __restrict__ bv,
    _Float16* __restrict__ Qh, _Float16* __restrict__ Kh,
    _Float16* __restrict__ Vt)
{
    const int lane = threadIdx.x & 31;
    const int wave = threadIdx.x >> 5;
    const int rowbase = (blockIdx.x * 8 + wave) * 16;   // 16 rows per wave
    const int r  = lane & 15;
    const int h  = lane >> 4;
    const int n  = lane & 15;
    const int kh = h * 16;

    // A-operand: rows of x, f32 -> f16, WMMA 16-bit A layout (4 K-steps of 32)
    v16h a[4];
    const float* xrow = x + (size_t)(rowbase + r) * EMBED;
#pragma unroll
    for (int ks = 0; ks < 4; ++ks) {
        const float* p0 = xrow + ks * 32 + 16 * h;       // K = 16h .. 16h+7
        const float* p1 = xrow + ks * 32 + 16 * h + 8;   // K = 16h+8 ..
#pragma unroll
        for (int j = 0; j < 8; ++j) {
            a[ks][j]     = (_Float16)p0[j];
            a[ks][8 + j] = (_Float16)p1[j];
        }
    }

#pragma unroll
    for (int nt = 0; nt < 8; ++nt) {                     // Q
        v8f acc = proj_acc(Wq, a, nt, n, kh);
        const float bias = bq[nt * 16 + n];
#pragma unroll
        for (int rr = 0; rr < 8; ++rr)
            Qh[(size_t)(rowbase + rr + 8 * h) * EMBED + nt * 16 + n] =
                (_Float16)(acc[rr] + bias);
    }
#pragma unroll
    for (int nt = 0; nt < 8; ++nt) {                     // K
        v8f acc = proj_acc(Wk, a, nt, n, kh);
        const float bias = bk[nt * 16 + n];
#pragma unroll
        for (int rr = 0; rr < 8; ++rr)
            Kh[(size_t)(rowbase + rr + 8 * h) * EMBED + nt * 16 + n] =
                (_Float16)(acc[rr] + bias);
    }
#pragma unroll
    for (int nt = 0; nt < 8; ++nt) {                     // V -> transposed
        v8f acc = proj_acc(Wv, a, nt, n, kh);
        const float bias = bv[nt * 16 + n];
#pragma unroll
        for (int rr = 0; rr < 8; ++rr) {
            const int row = rowbase + rr + 8 * h;
            const int bb  = row >> 12;                   // / 4096
            const int ss  = row & 4095;
            Vt[((size_t)bb * EMBED + nt * 16 + n) * SEQ + ss] =
                (_Float16)(acc[rr] + bias);
        }
    }
}

// ---------------------------------------------------------------------------
// Kernel 2: flash attention, block-cooperative K/V staging.
//   8 waves/block share one batch; each 32-key block of K (8KB) and Vt (8KB)
//   is async-copied into LDS once (double-buffered), then all waves consume.
// ---------------------------------------------------------------------------
__device__ __forceinline__ void issue_copy(const _Float16* __restrict__ Kb,
                                           const _Float16* __restrict__ Vb,
                                           _Float16* kdst, _Float16* vdst,
                                           int kk, int tid) {
    // K rows kk..kk+31 are one contiguous 8KB region
    const char* kg = (const char*)(Kb + (size_t)kk * EMBED);
    char*       kl = (char*)kdst;
#pragma unroll
    for (int c = 0; c < 2; ++c) {
        const int off = (tid + c * 256) * 16;            // 2 x 4KB halves
        copy16(kg + off, kl + off);
    }
    // Vt slice: 128 rows x 64B, global row stride 8KB
#pragma unroll
    for (int c = 0; c < 2; ++c) {
        const int v   = tid + c * 256;                   // 0..511
        const int row = v >> 2, part = v & 3;
        copy16((const char*)(Vb + (size_t)row * SEQ + kk) + part * 16,
               (char*)vdst + row * 64 + part * 16);
    }
}

__global__ __launch_bounds__(256) void flash_attn_kernel(
    const _Float16* __restrict__ Qh, const _Float16* __restrict__ Kh,
    const _Float16* __restrict__ Vt, float* __restrict__ out)
{
    __shared__ __align__(32) _Float16 Ksh[2][32 * EMBED];   // 2 x 8KB
    __shared__ __align__(32) _Float16 Vsh[2][EMBED * 32];   // 2 x 8KB
    __shared__ __align__(32) _Float16 Psh[8][16 * 32];      // 8KB P staging

    const int tid   = threadIdx.x;
    const int lane  = tid & 31;
    const int wave  = tid >> 5;
    const int tile  = blockIdx.x * 8 + wave;             // 0..1023
    const int b     = tile >> 8;                         // batch (uniform/block)
    const int qbase = (tile & 255) * 16;
    const int r  = lane & 15;
    const int h  = lane >> 4;
    const int n  = lane & 15;
    const int kh = h * 16;

    const _Float16* Qb = Qh + (size_t)b * SEQ * EMBED;
    const _Float16* Kb = Kh + (size_t)b * SEQ * EMBED;
    const _Float16* Vb = Vt + (size_t)b * EMBED * SEQ;

    // Q tile in A layout, registers for the whole pass
    v16h qa[4];
    const _Float16* qrow = Qb + (size_t)(qbase + r) * EMBED;
#pragma unroll
    for (int ks = 0; ks < 4; ++ks) {
        v8h lo = *(const v8h*)(qrow + ks * 32 + 16 * h);
        v8h hi = *(const v8h*)(qrow + ks * 32 + 16 * h + 8);
#pragma unroll
        for (int j = 0; j < 8; ++j) { qa[ks][j] = lo[j]; qa[ks][8 + j] = hi[j]; }
    }

    v8f o[8] = {};
    float mrow[8], lrow[8];
#pragma unroll
    for (int i = 0; i < 8; ++i) { mrow[i] = -1e30f; lrow[i] = 0.0f; }

    _Float16* lp = &Psh[wave][0];

    issue_copy(Kb, Vb, Ksh[0], Vsh[0], 0, tid);          // prologue

    for (int it = 0; it < SEQ / 32; ++it) {
        const int kk  = it * 32;
        const int cur = it & 1;
        if (it + 1 < SEQ / 32) {
            issue_copy(Kb, Vb, Ksh[cur ^ 1], Vsh[cur ^ 1], kk + 32, tid);
#ifdef HAVE_ASYNC_LDS
            asm volatile("s_wait_asynccnt 4" ::: "memory");  // current block done
#endif
        } else {
#ifdef HAVE_ASYNC_LDS
            asm volatile("s_wait_asynccnt 0" ::: "memory");
#endif
        }
        __syncthreads();                                 // all waves' copies done

        const _Float16* ksh = Ksh[cur];
        const _Float16* vsh = Vsh[cur];

        // --- S = Q * K^T for 32 keys: two 16x16 f32 tiles ------------------
        v8f s0 = {}, s1 = {};
#pragma unroll
        for (int ks = 0; ks < 4; ++ks) {
            v16h kb0 = *(const v16h*)(ksh + (size_t)n        * EMBED + ks * 32 + kh);
            v16h kb1 = *(const v16h*)(ksh + (size_t)(16 + n) * EMBED + ks * 32 + kh);
            s0 = wmma16(qa[ks], kb0, s0);
            s1 = wmma16(qa[ks], kb1, s1);
        }

        // --- online softmax (rows span 16 lanes in C layout) ---------------
        float corr[8];
#pragma unroll
        for (int i = 0; i < 8; ++i) {
            float mx = fmaxf(s0[i], s1[i]);
#pragma unroll
            for (int msk = 1; msk < 16; msk <<= 1)
                mx = fmaxf(mx, __shfl_xor(mx, msk));
            mx *= SCALE;
            const float mnew = fmaxf(mrow[i], mx);
            corr[i] = __expf(mrow[i] - mnew);
            mrow[i] = mnew;
        }
#pragma unroll
        for (int i = 0; i < 8; ++i) {
            const float e0 = __expf(s0[i] * SCALE - mrow[i]);
            const float e1 = __expf(s1[i] * SCALE - mrow[i]);
            lp[(i + 8 * h) * 32 + n]      = (_Float16)e0;
            lp[(i + 8 * h) * 32 + 16 + n] = (_Float16)e1;
            float rs = e0 + e1;
#pragma unroll
            for (int msk = 1; msk < 16; msk <<= 1)
                rs += __shfl_xor(rs, msk);
            lrow[i] = lrow[i] * corr[i] + rs;
        }

        asm volatile("s_wait_dscnt 0" ::: "memory");     // P writes visible

        // --- reload P in A layout (16x32 f16) ------------------------------
        v16h pa;
        {
            const _Float16* pr = lp + r * 32;
            v8h lo = *(const v8h*)(pr + 16 * h);
            v8h hi = *(const v8h*)(pr + 16 * h + 8);
#pragma unroll
            for (int j = 0; j < 8; ++j) { pa[j] = lo[j]; pa[8 + j] = hi[j]; }
        }

        // --- O = O*corr + P @ V --------------------------------------------
#pragma unroll
        for (int t = 0; t < 8; ++t) {
#pragma unroll
            for (int i = 0; i < 8; ++i) o[t][i] *= corr[i];
            v16h vb = *(const v16h*)(vsh + (t * 16 + n) * 32 + kh);
            o[t] = wmma16(pa, vb, o[t]);
        }
        __syncthreads();                                 // before buffer reuse
    }

    // --- finalize: divide by softmax denominator, store fp32 ----------------
    float inv[8];
#pragma unroll
    for (int i = 0; i < 8; ++i) inv[i] = 1.0f / lrow[i];
    float* orow = out + ((size_t)b * SEQ + qbase) * EMBED;
#pragma unroll
    for (int t = 0; t < 8; ++t)
#pragma unroll
        for (int i = 0; i < 8; ++i)
            orow[(size_t)(i + 8 * h) * EMBED + t * 16 + n] = o[t][i] * inv[i];
}

// ---------------------------------------------------------------------------
extern "C" void kernel_launch(void* const* d_in, const int* in_sizes, int n_in,
                              void* d_out, int out_size, void* d_ws, size_t ws_size,
                              hipStream_t stream) {
    const float* x  = (const float*)d_in[0];
    const float* Wq = (const float*)d_in[1];
    const float* bq = (const float*)d_in[2];
    const float* Wk = (const float*)d_in[3];
    const float* bk = (const float*)d_in[4];
    const float* Wv = (const float*)d_in[5];
    const float* bv = (const float*)d_in[6];
    float* out = (float*)d_out;

    const size_t elems = (size_t)BATCH * SEQ * EMBED;    // 2M f16 each
    _Float16* Qh = (_Float16*)d_ws;
    _Float16* Kh = Qh + elems;
    _Float16* Vt = Kh + elems;

    qkv_proj_kernel<<<128, 256, 0, stream>>>(x, Wq, bq, Wk, bk, Wv, bv, Qh, Kh, Vt);
    flash_attn_kernel<<<128, 256, 0, stream>>>(Qh, Kh, Vt, out);
}
